// ModelGraph_39548058861638
// MI455X (gfx1250) — matrix operand order repaired and checked by reading, not compiled, not run on hardware
//
#include <hip/hip_runtime.h>
#include <cstdint>
#include <cstddef>

// ---------------- problem constants (from reference) ----------------
#define V_   10000
#define E_   256
#define H_   512
#define B_   64
#define LX_  128
#define LY_  128
#define START_ 1
#define G3H_ 1536   // 3*H

// ---------------- CDNA5 WMMA types ----------------
typedef __attribute__((ext_vector_type(16))) __bf16 bf16x16;
typedef __attribute__((ext_vector_type(8)))  float  f32x8;

static __device__ __forceinline__ f32x8 zero8() {
  f32x8 z;
#pragma unroll
  for (int i = 0; i < 8; ++i) z[i] = 0.0f;
  return z;
}

static __device__ __forceinline__ f32x8 wmma_bf16(bf16x16 a, bf16x16 b, f32x8 c) {
  // (neg_a, A, neg_b, B, c_mod, C, reuse_a, reuse_b)
  return __builtin_amdgcn_wmma_f32_16x16x32_bf16(false, a, false, b, (short)0, c,
                                                 false, false);
}

union FragU { bf16x16 v; uint4 q[2]; };

// A fragment (16x32, bf16, row-major source).
// rowptr must point at element [m = m0 + lane%16, k0] of the source.
static __device__ __forceinline__ bf16x16 load_frag_a(const uint16_t* rowptr, int lane) {
  const uint16_t* p = rowptr + ((lane >> 4) << 3);  // + 8*(lane/16)
  FragU u;
  u.q[0] = *(const uint4*)(p);        // K = 8g + 0..7
  u.q[1] = *(const uint4*)(p + 16);   // K = 8g + 16..23
  return u.v;
}

// B fragment (32x16, bf16) from W stored [N, K] row-major (ldk elements).
static __device__ __forceinline__ bf16x16 load_frag_b(const uint16_t* W, size_t n,
                                                      int ldk, int k0, int lane) {
  const uint16_t* p = W + n * (size_t)ldk + k0 + ((lane >> 4) << 3); // + 8*(lane/16)
  FragU u;
  u.q[0] = *(const uint4*)(p);        // K = 8g + 0..7
  u.q[1] = *(const uint4*)(p + 16);   // K = 8g + 16..23
  return u.v;
}

static __device__ __forceinline__ uint16_t f2bf(float f) {
  uint32_t u = __builtin_bit_cast(uint32_t, f);
  u += 0x7FFFu + ((u >> 16) & 1u);    // round-to-nearest-even
  return (uint16_t)(u >> 16);
}

// ---------------- kernel: fp32 -> bf16 (optional relu) ----------------
__global__ __launch_bounds__(256) void k_f32_to_bf16(const float* __restrict__ src,
                                                     uint16_t* __restrict__ dst,
                                                     int n, int relu) {
  int i = blockIdx.x * 256 + threadIdx.x;
  if (i < n) {
    float f = src[i];
    if (relu) f = fmaxf(f, 0.0f);
    dst[i] = f2bf(f);
  }
}

// ---------------- kernel: GI = emb[idx] @ Wih^T + bih ----------------
// GI rows ordered [t*B + b]; token = xs[b, t] (encoder) or teacher-forced (decoder).
// Software-pipelined k-loop (double-buffered fragments).
__global__ __launch_bounds__(256) void k_gi_gemm(const int* __restrict__ tokens,   // [B, L]
                                                 const uint16_t* __restrict__ emb, // [V, E] bf16
                                                 const uint16_t* __restrict__ Wih, // [3H, E] bf16
                                                 const float* __restrict__ bih,    // [3H]
                                                 float* __restrict__ gi,           // [L*B, 3H]
                                                 int L, int is_dec) {
  const int tid  = threadIdx.x;
  const int lane = tid & 31, wave = tid >> 5;
  const int g    = lane >> 4, mr = lane & 15;
  const int m0   = (blockIdx.y * 8 + wave) * 16;
  const int n0   = blockIdx.x * 64;

  // per-lane gathered embedding row
  const int row = m0 + mr;
  const int t = row >> 6, b = row & 63;
  int tok;
  if (is_dec) {
    int tprev = (t == 0) ? 0 : (t - 1);
    int tk = tokens[b * L + tprev];
    tok = (t == 0) ? START_ : tk;
  } else {
    tok = tokens[b * L + t];
  }
  const uint16_t* arow = emb + (size_t)tok * E_;

  f32x8 acc[4];
#pragma unroll
  for (int i = 0; i < 4; ++i) acc[i] = zero8();

  bf16x16 a_c = load_frag_a(arow, lane);
  bf16x16 b_c[4], b_n[4];
#pragma unroll
  for (int i = 0; i < 4; ++i)
    b_c[i] = load_frag_b(Wih, (size_t)(n0 + i * 16 + mr), E_, 0, lane);

#pragma unroll
  for (int k0 = 0; k0 < E_; k0 += 32) {
    bf16x16 a_n;
    if (k0 + 32 < E_) {
      a_n = load_frag_a(arow + k0 + 32, lane);
#pragma unroll
      for (int i = 0; i < 4; ++i)
        b_n[i] = load_frag_b(Wih, (size_t)(n0 + i * 16 + mr), E_, k0 + 32, lane);
    }
#pragma unroll
    for (int i = 0; i < 4; ++i) acc[i] = wmma_bf16(a_c, b_c[i], acc[i]);
    a_c = a_n;
#pragma unroll
    for (int i = 0; i < 4; ++i) b_c[i] = b_n[i];
  }

#pragma unroll
  for (int i = 0; i < 4; ++i) {
    int n = n0 + i * 16 + mr;
    float bias = bih[n];
#pragma unroll
    for (int v = 0; v < 8; ++v) {
      int m = m0 + 8 * g + v;
      gi[(size_t)m * G3H_ + n] = acc[i][v] + bias;
    }
  }
}

// ---------------- kernel: GRU recurrence (one persistent workgroup) ----------------
// 512 threads = 16 waves. Wave w owns m_tile = w/4 (batch rows) and 8 j-tiles
// (w%4)*8 .. +7 (hidden cols). Per step: 3 gate accumulators per tile via WMMA
// from h_bf (LDS), B fragments double-buffered from L2, then fused gate math with
// fp32 h held in registers. Next step's GI slab is prefetched during the GEMM.
__global__ __launch_bounds__(512) void k_gru_seq(const float* __restrict__ gi,      // [L*64, 1536]
                                                 const uint16_t* __restrict__ Whh,  // [1536, 512] bf16
                                                 const float* __restrict__ bhh,     // [1536]
                                                 const float* __restrict__ h_init,  // [64,512] or null
                                                 float* __restrict__ h_final,       // [64,512] or null
                                                 uint16_t* __restrict__ hs,         // [L*64,512] bf16 or null
                                                 int L) {
  __shared__ uint16_t h_bf[B_ * H_];   // 64 KB

  const int tid  = threadIdx.x;
  const int lane = tid & 31, wave = tid >> 5;
  const int g    = lane >> 4, mr = lane & 15;
  const int m0   = (wave >> 2) * 16;
  const int jtb  = (wave & 3) * 8;     // first of 8 j-tiles

  float h_old[8][8];
#pragma unroll
  for (int i = 0; i < 8; ++i) {
    int col = (jtb + i) * 16 + mr;
#pragma unroll
    for (int v = 0; v < 8; ++v) {
      int brow = m0 + 8 * g + v;
      float h0 = h_init ? h_init[brow * H_ + col] : 0.0f;
      h_old[i][v] = h0;
      h_bf[brow * H_ + col] = f2bf(h0);
    }
  }
  __syncthreads();

  for (int t = 0; t < L; ++t) {
    // Pull next step's GI slab (384 KB) toward the caches while this step computes.
    if (t + 1 < L) {
      const char* base = (const char*)(gi + (size_t)(t + 1) * B_ * G3H_);
#pragma unroll
      for (int it = 0; it < (B_ * G3H_ * 4) / (512 * 64); ++it)
        __builtin_prefetch(base + (size_t)(it * 512 + tid) * 64, 0, 0);
    }

    float h_new[8][8];
#pragma unroll
    for (int i = 0; i < 8; ++i) {
      const int col = (jtb + i) * 16 + mr;
      f32x8 ar = zero8(), az = zero8(), an = zero8();
      const uint16_t* a_row = &h_bf[(m0 + mr) * H_];
      const size_t nr = (size_t)col, nz = (size_t)(H_ + col), nn2 = (size_t)(2 * H_ + col);

      bf16x16 br_c = load_frag_b(Whh, nr,  H_, 0, lane);
      bf16x16 bz_c = load_frag_b(Whh, nz,  H_, 0, lane);
      bf16x16 bn_c = load_frag_b(Whh, nn2, H_, 0, lane);
#pragma unroll
      for (int k0 = 0; k0 < H_; k0 += 32) {
        bf16x16 br_n, bz_n, bn_n;
        if (k0 + 32 < H_) {
          br_n = load_frag_b(Whh, nr,  H_, k0 + 32, lane);
          bz_n = load_frag_b(Whh, nz,  H_, k0 + 32, lane);
          bn_n = load_frag_b(Whh, nn2, H_, k0 + 32, lane);
        }
        bf16x16 a = load_frag_a(a_row + k0, lane);
        ar = wmma_bf16(a, br_c, ar);
        az = wmma_bf16(a, bz_c, az);
        an = wmma_bf16(a, bn_c, an);
        br_c = br_n; bz_c = bz_n; bn_c = bn_n;
      }

      float bh_r = bhh[col], bh_z = bhh[H_ + col], bh_n = bhh[2 * H_ + col];
#pragma unroll
      for (int v = 0; v < 8; ++v) {
        int brow = m0 + 8 * g + v;
        const float* gp = gi + ((size_t)t * B_ + brow) * G3H_ + col;
        float ir = gp[0], iz = gp[H_], inn = gp[2 * H_];
        float r  = 1.0f / (1.0f + expf(-(ir + ar[v] + bh_r)));
        float z  = 1.0f / (1.0f + expf(-(iz + az[v] + bh_z)));
        float nn = tanhf(inn + r * (an[v] + bh_n));
        h_new[i][v] = (1.0f - z) * nn + z * h_old[i][v];
      }
    }
    __syncthreads();   // all WMMA reads of h_bf done
#pragma unroll
    for (int i = 0; i < 8; ++i) {
      int col = (jtb + i) * 16 + mr;
#pragma unroll
      for (int v = 0; v < 8; ++v) {
        int brow = m0 + 8 * g + v;
        uint16_t hb = f2bf(h_new[i][v]);
        h_bf[brow * H_ + col] = hb;
        h_old[i][v] = h_new[i][v];
        if (hs) hs[((size_t)t * B_ + brow) * H_ + col] = hb;
      }
    }
    __syncthreads();   // h_bf writes visible before next step
  }

  if (h_final) {
#pragma unroll
    for (int i = 0; i < 8; ++i) {
      int col = (jtb + i) * 16 + mr;
#pragma unroll
      for (int v = 0; v < 8; ++v)
        h_final[(m0 + 8 * g + v) * H_ + col] = h_old[i][v];
    }
  }
}

// ---------------- kernel: logits = hs @ out_W^T + out_b, transposed store ----------------
__global__ __launch_bounds__(256) void k_proj(const uint16_t* __restrict__ hs,  // [8192, 512] bf16
                                              const uint16_t* __restrict__ Wo,  // [10000, 512] bf16
                                              const float* __restrict__ bo,     // [10000]
                                              float* __restrict__ out) {        // [64, 128, 10000]
  const int tid  = threadIdx.x;
  const int lane = tid & 31, wave = tid >> 5;
  const int g    = lane >> 4, mr = lane & 15;
  const int m0   = (blockIdx.y * 8 + wave) * 16;
  const int ntb  = blockIdx.x * 4;                 // n-tile base (16 cols per tile)
  const int NT   = V_ / 16;                        // 625 n-tiles

  const uint16_t* arow = hs + (size_t)(m0 + mr) * H_;

  f32x8 acc[4];
#pragma unroll
  for (int i = 0; i < 4; ++i) acc[i] = zero8();

  bf16x16 a_c = load_frag_a(arow, lane);
  bf16x16 b_c[4], b_n[4];
#pragma unroll
  for (int i = 0; i < 4; ++i)
    if (ntb + i < NT)
      b_c[i] = load_frag_b(Wo, (size_t)((ntb + i) * 16 + mr), H_, 0, lane);

#pragma unroll
  for (int k0 = 0; k0 < H_; k0 += 32) {
    bf16x16 a_n;
    if (k0 + 32 < H_) {
      a_n = load_frag_a(arow + k0 + 32, lane);
#pragma unroll
      for (int i = 0; i < 4; ++i)
        if (ntb + i < NT)
          b_n[i] = load_frag_b(Wo, (size_t)((ntb + i) * 16 + mr), H_, k0 + 32, lane);
    }
#pragma unroll
    for (int i = 0; i < 4; ++i)
      if (ntb + i < NT)                 // uniform branch: EXEC stays full for WMMA
        acc[i] = wmma_bf16(a_c, b_c[i], acc[i]);
    a_c = a_n;
#pragma unroll
    for (int i = 0; i < 4; ++i) b_c[i] = b_n[i];
  }

#pragma unroll
  for (int i = 0; i < 4; ++i) {
    if (ntb + i < NT) {
      int n = (ntb + i) * 16 + mr;
      float bias = bo[n];
#pragma unroll
      for (int v = 0; v < 8; ++v) {
        int m = m0 + 8 * g + v;
        int l = m >> 6, b = m & 63;
        out[((size_t)b * LY_ + l) * V_ + n] = acc[i][v] + bias;
      }
    }
  }
}

// ---------------- kernel: in-place row-wise log-softmax over V ----------------
__global__ __launch_bounds__(256) void k_logsoftmax(float* __restrict__ out) {
  __shared__ float red[256];
  float* p = out + (size_t)blockIdx.x * V_;
  const int tid = threadIdx.x;

  float m = -3.402823466e38f;
  for (int i = tid; i < V_; i += 256) m = fmaxf(m, p[i]);
  red[tid] = m; __syncthreads();
  for (int s = 128; s > 0; s >>= 1) {
    if (tid < s) red[tid] = fmaxf(red[tid], red[tid + s]);
    __syncthreads();
  }
  m = red[0]; __syncthreads();

  float sum = 0.0f;
  for (int i = tid; i < V_; i += 256) sum += expf(p[i] - m);
  red[tid] = sum; __syncthreads();
  for (int s = 128; s > 0; s >>= 1) {
    if (tid < s) red[tid] += red[tid + s];
    __syncthreads();
  }
  float lse = m + logf(red[0]);

  for (int i = tid; i < V_; i += 256) p[i] = p[i] - lse;
}

// ---------------- workspace layout (bytes) ----------------
static constexpr size_t SZ_GI   = (size_t)(LX_ * B_) * G3H_ * 4;   // 50,331,648
static constexpr size_t OFF_GI_ENC = 0;
static constexpr size_t OFF_GI_DEC = OFF_GI_ENC + SZ_GI;
static constexpr size_t OFF_HENC   = OFF_GI_DEC + SZ_GI;
static constexpr size_t OFF_HS     = OFF_HENC + (size_t)B_ * H_ * 4;
static constexpr size_t OFF_EEMB   = OFF_HS   + (size_t)(LY_ * B_) * H_ * 2;
static constexpr size_t OFF_DEMB   = OFF_EEMB + (size_t)V_ * E_ * 2;
static constexpr size_t OFF_EWIH   = OFF_DEMB + (size_t)V_ * E_ * 2;
static constexpr size_t OFF_DWIH   = OFF_EWIH + (size_t)G3H_ * E_ * 2;
static constexpr size_t OFF_EWHH   = OFF_DWIH + (size_t)G3H_ * E_ * 2;
static constexpr size_t OFF_DWHH   = OFF_EWHH + (size_t)G3H_ * H_ * 2;
static constexpr size_t OFF_OUTW   = OFF_DWHH + (size_t)G3H_ * H_ * 2;

extern "C" void kernel_launch(void* const* d_in, const int* in_sizes, int n_in,
                              void* d_out, int out_size, void* d_ws, size_t ws_size,
                              hipStream_t stream) {
  const int*   xs      = (const int*)d_in[0];
  const int*   ys      = (const int*)d_in[1];
  const float* enc_emb = (const float*)d_in[2];
  const float* enc_Wih = (const float*)d_in[3];
  const float* enc_Whh = (const float*)d_in[4];
  const float* enc_bih = (const float*)d_in[5];
  const float* enc_bhh = (const float*)d_in[6];
  const float* dec_emb = (const float*)d_in[7];
  const float* dec_Wih = (const float*)d_in[8];
  const float* dec_Whh = (const float*)d_in[9];
  const float* dec_bih = (const float*)d_in[10];
  const float* dec_bhh = (const float*)d_in[11];
  const float* out_W   = (const float*)d_in[12];
  const float* out_b   = (const float*)d_in[13];
  float* out = (float*)d_out;

  char* ws = (char*)d_ws;
  float*    gi_enc  = (float*)   (ws + OFF_GI_ENC);
  float*    gi_dec  = (float*)   (ws + OFF_GI_DEC);
  float*    h_enc   = (float*)   (ws + OFF_HENC);
  uint16_t* hs_bf   = (uint16_t*)(ws + OFF_HS);
  uint16_t* eemb_bf = (uint16_t*)(ws + OFF_EEMB);
  uint16_t* demb_bf = (uint16_t*)(ws + OFF_DEMB);
  uint16_t* ewih_bf = (uint16_t*)(ws + OFF_EWIH);
  uint16_t* dwih_bf = (uint16_t*)(ws + OFF_DWIH);
  uint16_t* ewhh_bf = (uint16_t*)(ws + OFF_EWHH);
  uint16_t* dwhh_bf = (uint16_t*)(ws + OFF_DWHH);
  uint16_t* outw_bf = (uint16_t*)(ws + OFF_OUTW);

  auto conv = [&](const float* s, uint16_t* d, int n, int relu) {
    k_f32_to_bf16<<<(n + 255) / 256, 256, 0, stream>>>(s, d, n, relu);
  };
  conv(enc_emb, eemb_bf, V_ * E_, 0);
  conv(dec_emb, demb_bf, V_ * E_, 1);   // relu folds into the table (elementwise)
  conv(enc_Wih, ewih_bf, G3H_ * E_, 0);
  conv(dec_Wih, dwih_bf, G3H_ * E_, 0);
  conv(enc_Whh, ewhh_bf, G3H_ * H_, 0);
  conv(dec_Whh, dwhh_bf, G3H_ * H_, 0);
  conv(out_W,   outw_bf, V_ * H_, 0);

  // input-projection GEMMs (parallel over all timesteps)
  dim3 gridGI(G3H_ / 64, (LX_ * B_) / (16 * 8));   // (24, 64)
  k_gi_gemm<<<gridGI, 256, 0, stream>>>(xs, eemb_bf, ewih_bf, enc_bih, gi_enc, LX_, 0);
  k_gi_gemm<<<gridGI, 256, 0, stream>>>(ys, demb_bf, dwih_bf, dec_bih, gi_dec, LY_, 1);

  // sequential recurrences (single persistent workgroup each)
  k_gru_seq<<<1, 512, 0, stream>>>(gi_enc, ewhh_bf, enc_bhh, nullptr, h_enc, nullptr, LX_);
  k_gru_seq<<<1, 512, 0, stream>>>(gi_dec, dwhh_bf, dec_bhh, h_enc, nullptr, hs_bf, LY_);

  // output projection -> d_out (transposed to [B, LY, V]), then in-place log-softmax
  dim3 gridP((V_ / 16 + 3) / 4, (LY_ * B_) / (16 * 8));  // (157, 64)
  k_proj<<<gridP, 256, 0, stream>>>(hs_bf, outw_bf, out_b, out);
  k_logsoftmax<<<LY_ * B_, 256, 0, stream>>>(out);
}
